// LinearAttention3D_69217692942746
// MI455X (gfx1250) — matrix operand order repaired and checked by reading, not compiled, not run on hardware
//
#include <hip/hip_runtime.h>

#define HEADS 8
#define DH    64
#define DIMQ  512
#define CIN   256
#define F3    1536
#define NTOK  32768
#define BATCH 2
#define TOKS  (BATCH * NTOK)
#define EPSLN 1e-5f

typedef __attribute__((ext_vector_type(16))) __bf16 v16bf;
typedef __attribute__((ext_vector_type(8)))  float  v8f;

union BF16Frag { v16bf v; unsigned short u[16]; };
union F32Acc   { v8f   v; float f[8]; };

#if __has_builtin(__builtin_amdgcn_global_load_async_to_lds_b128)
#define HAS_ASYNC_LDS 1
typedef int v4i_vs __attribute__((vector_size(16)));
typedef __attribute__((address_space(3))) v4i_vs* lds_v4i_ptr;
#else
#define HAS_ASYNC_LDS 0
#endif

__device__ __forceinline__ void wait_asynccnt0() {
#if __has_builtin(__builtin_amdgcn_s_wait_asynccnt)
    __builtin_amdgcn_s_wait_asynccnt(0);
#else
    asm volatile("s_wait_asynccnt 0x0" ::: "memory");
#endif
}

__device__ __forceinline__ unsigned short f2bf(float x) {
    union { float f; unsigned u; } c; c.f = x;
    unsigned u = c.u;
    return (unsigned short)((u + 0x7FFFu + ((u >> 16) & 1u)) >> 16);
}

__device__ __forceinline__ v8f wmma_bf16(v16bf a, v16bf b, v8f c) {
    return __builtin_amdgcn_wmma_f32_16x16x32_bf16(false, a, false, b, (short)0, c, false, false);
}

// K index for element e of a 16x32 bf16 A fragment (ISA 7.12.2 layout).
__device__ __forceinline__ int a_kidx(int e, int hi) {
    int v = e >> 1, p = e & 1;
    int k = (v < 4) ? (2 * v + p) : (16 + 2 * (v - 4) + p);
    return k + (hi ? 8 : 0);
}
// K index for element e of a 32x16 bf16 B fragment.
__device__ __forceinline__ int b_kidx(int e, int hi) {
    int v = e >> 1, p = e & 1;
    int k = (v < 4) ? (2 * v + p) : (16 + 2 * (v - 4) + p);
    return k + (hi ? 8 : 0);
}

// ---------------------------------------------------------------- converts
__global__ void cvt_f32_bf16(const float* __restrict__ src,
                             unsigned short* __restrict__ dst, int n) {
    int i = blockIdx.x * blockDim.x + threadIdx.x;
    int stride = gridDim.x * blockDim.x;
    for (; i < n; i += stride) dst[i] = f2bf(src[i]);
}

__global__ void zero_f32(float* __restrict__ p, int n) {
    int i = blockIdx.x * blockDim.x + threadIdx.x;
    int stride = gridDim.x * blockDim.x;
    for (; i < n; i += stride) p[i] = 0.0f;
}

// ---------------------------------------------------------------- KV GEMM
// kv[tok][0:512] = K proj, kv[tok][512:1024] = V proj  (fp32)
__global__ __launch_bounds__(256) void kv_gemm(const unsigned short* __restrict__ xb,
                                               const unsigned short* __restrict__ wqkvb,
                                               float* __restrict__ kv) {
    __shared__ __align__(16) unsigned short As[128][40];  // row stride 80B (16B aligned)
    __shared__ __align__(16) unsigned short Bs[128][40];
    const int tid = threadIdx.x;
    const int lane = tid & 31, wave = tid >> 5;
    const int Mw = wave >> 2, Nw = wave & 3;
    const int hi = lane >> 4, l15 = lane & 15;
    const int m0 = blockIdx.x * 128;
    const int n0 = blockIdx.y * 128;

    F32Acc acc[4][2];
#pragma unroll
    for (int mt = 0; mt < 4; ++mt)
#pragma unroll
        for (int nt = 0; nt < 2; ++nt)
#pragma unroll
            for (int r = 0; r < 8; ++r) acc[mt][nt].f[r] = 0.0f;

    for (int kk = 0; kk < CIN; kk += 32) {
        __builtin_prefetch(&wqkvb[(size_t)(kk + 32) * F3 + 512 + n0], 0, 3);
#if HAS_ASYNC_LDS
        // A tile 128x32 bf16 = 128 rows x 64B, async DMA to LDS: 2 x 16B per lane
#pragma unroll
        for (int i = 0; i < 2; ++i) {
            int c16 = tid + 256 * i;                    // 16B-chunk index, 0..511
            int m = c16 >> 2;                           // row 0..127
            int ck = (c16 & 3) * 8;                     // col (ushorts) 0/8/16/24
            const unsigned short* g = &xb[(size_t)(m0 + m) * CIN + kk + ck];
            __builtin_amdgcn_global_load_async_to_lds_b128(
                (v4i_vs*)g,
                (lds_v4i_ptr)(unsigned)(unsigned long long)&As[m][ck],
                0, 0);
        }
#else
#pragma unroll
        for (int i = 0; i < 16; ++i) {                  // A tile 128x32
            int e = tid + 256 * i;
            int m = e >> 5, c = e & 31;
            As[m][c] = xb[(size_t)(m0 + m) * CIN + kk + c];
        }
#endif
#pragma unroll
        for (int i = 0; i < 16; ++i) {                  // B tile 32x128 (store N-major)
            int e = tid + 256 * i;
            int k = e >> 7, n = e & 127;
            Bs[n][k] = wqkvb[(size_t)(kk + k) * F3 + 512 + n0 + n];
        }
#if HAS_ASYNC_LDS
        wait_asynccnt0();
#endif
        __syncthreads();

        BF16Frag afr[4];
#pragma unroll
        for (int mt = 0; mt < 4; ++mt)
#pragma unroll
            for (int e = 0; e < 16; ++e)
                afr[mt].u[e] = As[Mw * 64 + mt * 16 + l15][a_kidx(e, hi)];
        BF16Frag bfr[2];
#pragma unroll
        for (int nt = 0; nt < 2; ++nt)
#pragma unroll
            for (int e = 0; e < 16; ++e)
                bfr[nt].u[e] = Bs[Nw * 32 + nt * 16 + l15][b_kidx(e, hi)];
#pragma unroll
        for (int mt = 0; mt < 4; ++mt)
#pragma unroll
            for (int nt = 0; nt < 2; ++nt)
                acc[mt][nt].v = wmma_bf16(afr[mt].v, bfr[nt].v, acc[mt][nt].v);
        __syncthreads();
    }
#pragma unroll
    for (int mt = 0; mt < 4; ++mt)
#pragma unroll
        for (int nt = 0; nt < 2; ++nt)
#pragma unroll
            for (int r = 0; r < 8; ++r) {
                int m = m0 + Mw * 64 + mt * 16 + r + hi * 8;
                int n = n0 + Nw * 32 + nt * 16 + l15;
                kv[(size_t)m * 1024 + n] = acc[mt][nt].f[r];
            }
}

// ---------------------------------------------------------------- k-softmax max
__global__ __launch_bounds__(256) void kmax_partial(const float* __restrict__ kv,
                                                    float* __restrict__ pmax) {
    const int chunk = blockIdx.x;           // 256 chunks of 256 tokens
    const int tok0 = chunk * 256;
    const int c1 = threadIdx.x, c2 = threadIdx.x + 256;
    float m1 = -3.402823466e38f, m2 = -3.402823466e38f;
    for (int n = 0; n < 256; ++n) {
        const float* row = kv + (size_t)(tok0 + n) * 1024;
        m1 = fmaxf(m1, row[c1]);
        m2 = fmaxf(m2, row[c2]);
    }
    pmax[(size_t)chunk * 512 + c1] = m1;
    pmax[(size_t)chunk * 512 + c2] = m2;
}

__global__ void kmax_reduce(const float* __restrict__ pmax, float* __restrict__ kmax) {
    int gid = blockIdx.x * blockDim.x + threadIdx.x;   // 1024 = B*512
    int b = gid >> 9, c = gid & 511;
    float m = -3.402823466e38f;
    for (int j = 0; j < 128; ++j)
        m = fmaxf(m, pmax[(size_t)(b * 128 + j) * 512 + c]);
    kmax[gid] = m;
}

// ---------------------------------------------------------------- context: ctx += E^T V
__global__ __launch_bounds__(128) void context_kernel(const float* __restrict__ kv,
                                                      const float* __restrict__ kmax,
                                                      float* __restrict__ ksum,
                                                      float* __restrict__ ctx) {
    __shared__ float e_lds[32][68];
    __shared__ float v_lds[32][68];
    const int chunk = blockIdx.x;           // 0..31 (1024 tokens each)
    const int h = blockIdx.y;
    const int b = blockIdx.z;
    const int tid = threadIdx.x;
    const int lane = tid & 31, wave = tid >> 5;   // 4 waves, each owns a 16-row d strip
    const int hi = lane >> 4, l15 = lane & 15;
    const int ch = tid & 63;
    const float km = kmax[b * 512 + h * 64 + ch];
    const size_t base = ((size_t)b * NTOK + (size_t)chunk * 1024) * 1024 + h * 64;

    F32Acc acc[4];
#pragma unroll
    for (int nt = 0; nt < 4; ++nt)
#pragma unroll
        for (int r = 0; r < 8; ++r) acc[nt].f[r] = 0.0f;
    float ksum_part = 0.0f;

    for (int step = 0; step < 32; ++step) {
        const size_t tb = base + (size_t)step * 32 * 1024;
#pragma unroll
        for (int i = 0; i < 16; ++i) {
            int e = tid + 128 * i;
            int t = e >> 6;                  // token 0..31; channel == tid&63
            float kval = kv[tb + (size_t)t * 1024 + ch];
            float ev = __expf(kval - km);
            e_lds[t][ch] = ev;
            ksum_part += ev;
            v_lds[t][ch] = kv[tb + (size_t)t * 1024 + 512 + ch];
        }
        __syncthreads();
        BF16Frag afr;                        // A = E^T : M=d (16), K=token (32)
#pragma unroll
        for (int e = 0; e < 16; ++e)
            afr.u[e] = f2bf(e_lds[a_kidx(e, hi)][wave * 16 + l15]);
        BF16Frag bfr[4];                     // B = V : K=token (32), N=e (16)
#pragma unroll
        for (int nt = 0; nt < 4; ++nt)
#pragma unroll
            for (int e = 0; e < 16; ++e)
                bfr[nt].u[e] = f2bf(v_lds[b_kidx(e, hi)][nt * 16 + l15]);
#pragma unroll
        for (int nt = 0; nt < 4; ++nt)
            acc[nt].v = wmma_bf16(afr.v, bfr[nt].v, acc[nt].v);
        __syncthreads();
    }
#pragma unroll
    for (int nt = 0; nt < 4; ++nt)
#pragma unroll
        for (int r = 0; r < 8; ++r) {
            int d = wave * 16 + r + hi * 8;
            int e = nt * 16 + l15;
            atomicAdd(&ctx[(((size_t)b * HEADS + h) * 64 + d) * 64 + e], acc[nt].f[r]);
        }
    atomicAdd(&ksum[b * 512 + h * 64 + ch], ksum_part);
}

__global__ void ctx_norm(float* __restrict__ ctx, const float* __restrict__ ksum) {
    int idx = blockIdx.x * 256 + threadIdx.x;   // 65536
    int d = (idx >> 6) & 63;
    int h = (idx >> 12) & 7;
    int b = idx >> 15;
    float s = ksum[b * 512 + h * 64 + d];
    ctx[idx] = ctx[idx] / (s * (float)NTOK);
}

// ---------------------------------------------------------------- fused Q / out / proj / LN
__global__ __launch_bounds__(256) void final_kernel(const unsigned short* __restrict__ xb,
                                                    const unsigned short* __restrict__ wqkvb,
                                                    const unsigned short* __restrict__ woutb,
                                                    const float* __restrict__ ctx,
                                                    const float* __restrict__ b_out,
                                                    const float* __restrict__ ln_scale,
                                                    float* __restrict__ out) {
    union Sm {
        unsigned short xs[32][264];          // phase 1-2: X tile (32x256 bf16)
        unsigned short qs[8][32][68];        // phase 4-5: softmaxed Q per head
        unsigned short os[32][520];          // phase 5-6: attention out (32x512)
    };
    __shared__ Sm sm;
    __shared__ float rowsum[32];
    __shared__ float rowsq[32];

    const int tid = threadIdx.x;
    const int lane = tid & 31, w = tid >> 5;   // wave w == head w for phases 2-5
    const int hi = lane >> 4, l15 = lane & 15;
    const int tok0 = blockIdx.x * 32;
    const int b = tok0 >> 15;

    if (tid < 32) { rowsum[tid] = 0.0f; rowsq[tid] = 0.0f; }
#pragma unroll
    for (int i = 0; i < 32; ++i) {             // load X tile
        int e = tid + 256 * i;
        int m = e >> 8, c = e & 255;
        sm.xs[m][c] = xb[(size_t)(tok0 + m) * CIN + c];
    }
    __syncthreads();

    // ---- phase 2: Q = X @ Wq[:, head cols]
    F32Acc q[2][4];
#pragma unroll
    for (int mt = 0; mt < 2; ++mt)
#pragma unroll
        for (int nt = 0; nt < 4; ++nt)
#pragma unroll
            for (int r = 0; r < 8; ++r) q[mt][nt].f[r] = 0.0f;

    for (int kk = 0; kk < CIN; kk += 32) {
        __builtin_prefetch(&wqkvb[(size_t)(kk + 32) * F3 + w * 64], 0, 3);
        BF16Frag afr[2];
#pragma unroll
        for (int mt = 0; mt < 2; ++mt)
#pragma unroll
            for (int e = 0; e < 16; ++e)
                afr[mt].u[e] = sm.xs[mt * 16 + l15][kk + a_kidx(e, hi)];
        BF16Frag bfr[4];
#pragma unroll
        for (int nt = 0; nt < 4; ++nt) {
            int col = w * 64 + nt * 16 + l15;
#pragma unroll
            for (int e = 0; e < 16; ++e) {
                int k = kk + b_kidx(e, hi);
                bfr[nt].u[e] = wqkvb[(size_t)k * F3 + col];
            }
        }
#pragma unroll
        for (int mt = 0; mt < 2; ++mt)
#pragma unroll
            for (int nt = 0; nt < 4; ++nt)
                q[mt][nt].v = wmma_bf16(afr[mt].v, bfr[nt].v, q[mt][nt].v);
    }

    // ---- phase 3: row softmax over 64 head channels, scale 1/sqrt(64)
#pragma unroll
    for (int mt = 0; mt < 2; ++mt)
#pragma unroll
        for (int r = 0; r < 8; ++r) {
            float v0 = q[mt][0].f[r], v1 = q[mt][1].f[r];
            float v2 = q[mt][2].f[r], v3 = q[mt][3].f[r];
            float mx = fmaxf(fmaxf(v0, v1), fmaxf(v2, v3));
            for (int sh = 1; sh < 16; sh <<= 1) mx = fmaxf(mx, __shfl_xor(mx, sh, 32));
            v0 = __expf(v0 - mx); v1 = __expf(v1 - mx);
            v2 = __expf(v2 - mx); v3 = __expf(v3 - mx);
            float s = v0 + v1 + v2 + v3;
            for (int sh = 1; sh < 16; sh <<= 1) s += __shfl_xor(s, sh, 32);
            float inv = 1.0f / (s * 8.0f);
            q[mt][0].f[r] = v0 * inv; q[mt][1].f[r] = v1 * inv;
            q[mt][2].f[r] = v2 * inv; q[mt][3].f[r] = v3 * inv;
        }
    __syncthreads();                            // xs dead for all waves
    // ---- phase 4: stage Q (bf16) to LDS
#pragma unroll
    for (int mt = 0; mt < 2; ++mt)
#pragma unroll
        for (int nt = 0; nt < 4; ++nt)
#pragma unroll
            for (int r = 0; r < 8; ++r) {
                int row = mt * 16 + r + hi * 8;
                int col = nt * 16 + l15;
                sm.qs[w][row][col] = f2bf(q[mt][nt].f[r]);
            }
    __syncthreads();

    // ---- phase 5: OUT = Qsm @ ctx_head
    F32Acc o[2][4];
#pragma unroll
    for (int mt = 0; mt < 2; ++mt)
#pragma unroll
        for (int nt = 0; nt < 4; ++nt)
#pragma unroll
            for (int r = 0; r < 8; ++r) o[mt][nt].f[r] = 0.0f;
    const float* ctxh = ctx + ((size_t)b * HEADS + w) * 64 * 64;
    for (int kk = 0; kk < 64; kk += 32) {
        BF16Frag afr[2];
#pragma unroll
        for (int mt = 0; mt < 2; ++mt)
#pragma unroll
            for (int e = 0; e < 16; ++e)
                afr[mt].u[e] = sm.qs[w][mt * 16 + l15][kk + a_kidx(e, hi)];
        BF16Frag bfr[4];
#pragma unroll
        for (int nt = 0; nt < 4; ++nt)
#pragma unroll
            for (int e = 0; e < 16; ++e) {
                int k = kk + b_kidx(e, hi);
                bfr[nt].u[e] = f2bf(ctxh[k * 64 + nt * 16 + l15]);
            }
#pragma unroll
        for (int mt = 0; mt < 2; ++mt)
#pragma unroll
            for (int nt = 0; nt < 4; ++nt)
                o[mt][nt].v = wmma_bf16(afr[mt].v, bfr[nt].v, o[mt][nt].v);
    }
    __syncthreads();                            // qs dead for all waves
#pragma unroll
    for (int mt = 0; mt < 2; ++mt)
#pragma unroll
        for (int nt = 0; nt < 4; ++nt)
#pragma unroll
            for (int r = 0; r < 8; ++r) {
                int row = mt * 16 + r + hi * 8;
                int col = w * 64 + nt * 16 + l15;
                sm.os[row][col] = f2bf(o[mt][nt].f[r]);
            }
    __syncthreads();

    // ---- phase 6: Y = OUT @ Wout (+ bias), wave w owns cols [w*32, w*32+32)
    F32Acc y[2][2];
#pragma unroll
    for (int mt = 0; mt < 2; ++mt)
#pragma unroll
        for (int nt = 0; nt < 2; ++nt)
#pragma unroll
            for (int r = 0; r < 8; ++r) y[mt][nt].f[r] = 0.0f;
    for (int kk = 0; kk < DIMQ; kk += 32) {
        BF16Frag afr[2];
#pragma unroll
        for (int mt = 0; mt < 2; ++mt)
#pragma unroll
            for (int e = 0; e < 16; ++e)
                afr[mt].u[e] = sm.os[mt * 16 + l15][kk + a_kidx(e, hi)];
        BF16Frag bfr[2];
#pragma unroll
        for (int nt = 0; nt < 2; ++nt)
#pragma unroll
            for (int e = 0; e < 16; ++e) {
                int k = kk + b_kidx(e, hi);
                bfr[nt].u[e] = woutb[(size_t)k * CIN + w * 32 + nt * 16 + l15];
            }
#pragma unroll
        for (int mt = 0; mt < 2; ++mt)
#pragma unroll
            for (int nt = 0; nt < 2; ++nt)
                y[mt][nt].v = wmma_bf16(afr[mt].v, bfr[nt].v, y[mt][nt].v);
    }
#pragma unroll
    for (int mt = 0; mt < 2; ++mt)
#pragma unroll
        for (int nt = 0; nt < 2; ++nt)
#pragma unroll
            for (int r = 0; r < 8; ++r) {
                int col = w * 32 + nt * 16 + l15;
                y[mt][nt].f[r] += b_out[col];
            }
    // ---- phase 7: LayerNorm over 256 channels (cross-wave via LDS)
#pragma unroll
    for (int mt = 0; mt < 2; ++mt)
#pragma unroll
        for (int r = 0; r < 8; ++r) {
            float s = y[mt][0].f[r] + y[mt][1].f[r];
            float sq = y[mt][0].f[r] * y[mt][0].f[r] + y[mt][1].f[r] * y[mt][1].f[r];
            for (int sh = 1; sh < 16; sh <<= 1) {
                s += __shfl_xor(s, sh, 32);
                sq += __shfl_xor(sq, sh, 32);
            }
            if (l15 == 0) {
                int row = mt * 16 + r + hi * 8;
                atomicAdd(&rowsum[row], s);
                atomicAdd(&rowsq[row], sq);
            }
        }
    __syncthreads();
#pragma unroll
    for (int mt = 0; mt < 2; ++mt)
#pragma unroll
        for (int nt = 0; nt < 2; ++nt)
#pragma unroll
            for (int r = 0; r < 8; ++r) {
                int row = mt * 16 + r + hi * 8;
                float mu = rowsum[row] * (1.0f / 256.0f);
                float var = rowsq[row] * (1.0f / 256.0f) - mu * mu;
                float rstd = rsqrtf(var + EPSLN);
                int col = w * 32 + nt * 16 + l15;
                out[(size_t)(tok0 + row) * CIN + col] =
                    (y[mt][nt].f[r] - mu) * rstd * ln_scale[col];
            }
}

// ---------------------------------------------------------------- launch
extern "C" void kernel_launch(void* const* d_in, const int* in_sizes, int n_in,
                              void* d_out, int out_size, void* d_ws, size_t ws_size,
                              hipStream_t stream) {
    const float* x        = (const float*)d_in[0];
    const float* w_qkv    = (const float*)d_in[1];
    const float* w_out    = (const float*)d_in[2];
    const float* b_out    = (const float*)d_in[3];
    const float* ln_scale = (const float*)d_in[4];
    float* out = (float*)d_out;

    char* ws = (char*)d_ws;
    auto carve = [&](size_t bytes) {
        char* p = ws;
        ws += (bytes + 255) & ~(size_t)255;
        return (void*)p;
    };
    unsigned short* xb    = (unsigned short*)carve((size_t)TOKS * CIN * 2);
    unsigned short* wqkvb = (unsigned short*)carve((size_t)CIN * F3 * 2);
    unsigned short* woutb = (unsigned short*)carve((size_t)DIMQ * CIN * 2);
    float* kmax = (float*)carve((size_t)BATCH * 512 * 4);
    float* ksum = (float*)carve((size_t)BATCH * 512 * 4);
    float* ctx  = (float*)carve((size_t)BATCH * HEADS * 64 * 64 * 4);
    float* pmax = (float*)carve((size_t)256 * 512 * 4);
    float* kv   = (float*)carve((size_t)TOKS * 1024 * 4);

    cvt_f32_bf16<<<2048, 256, 0, stream>>>(x, xb, TOKS * CIN);
    cvt_f32_bf16<<<512, 256, 0, stream>>>(w_qkv, wqkvb, CIN * F3);
    cvt_f32_bf16<<<256, 256, 0, stream>>>(w_out, woutb, DIMQ * CIN);
    zero_f32<<<64, 256, 0, stream>>>(ctx, BATCH * HEADS * 64 * 64);
    zero_f32<<<4, 256, 0, stream>>>(ksum, BATCH * 512);

    kv_gemm<<<dim3(TOKS / 128, 8), 256, 0, stream>>>(xb, wqkvb, kv);
    kmax_partial<<<TOKS / 256, 256, 0, stream>>>(kv, pmax);
    kmax_reduce<<<4, 256, 0, stream>>>(pmax, kmax);
    context_kernel<<<dim3(32, HEADS, BATCH), 128, 0, stream>>>(kv, kmax, ksum, ctx);
    ctx_norm<<<256, 256, 0, stream>>>(ctx, ksum);
    final_kernel<<<TOKS / 32, 256, 0, stream>>>(xb, wqkvb, woutb, ctx, b_out,
                                                ln_scale, out);
}
